// Attention_2207613190299
// MI455X (gfx1250) — matrix operand order repaired and checked by reading, not compiled, not run on hardware
//
#include <hip/hip_runtime.h>
#include <hip/hip_bf16.h>

typedef __attribute__((ext_vector_type(16))) _Float16 v16h;
typedef __attribute__((ext_vector_type(8)))  _Float16 v8h;
typedef __attribute__((ext_vector_type(8)))  float    v8f;

#define EPS 1e-5f
#define ALPHA 100.0f

// ---------------------------------------------------------------------------
// 1x1 conv as GEMM: out[n,co,p] = b[co] + sum_ci W[co,ci]*in[n,ci,p]
__global__ void conv1x1_kernel(const float* __restrict__ in,
                               const float* __restrict__ W,
                               const float* __restrict__ bias,
                               float* __restrict__ out) {
    int tid = blockIdx.x * 256 + threadIdx.x;        // (n*64+co)*4096 + p
    int p  = tid & 4095;
    int co = (tid >> 12) & 63;
    int n  = tid >> 18;
    const float* src = in + ((size_t)n << 20) + p;
    const float* w   = W + co * 256;
    float acc = bias[co];
    #pragma unroll 4
    for (int k = 0; k < 256; ++k)
        acc += w[k] * src[(size_t)k << 12];
    out[tid] = acc;
}

// ---------------------------------------------------------------------------
// Instance-norm (+ leaky-relu 0.2), one block per (n,c). Safe in-place.
__global__ void inorm_leaky_kernel(const float* __restrict__ in,
                                   float* __restrict__ out, int HW) {
    __shared__ float sh1[256], sh2[256];
    const float* x = in + (size_t)blockIdx.x * HW;
    float* y = out + (size_t)blockIdx.x * HW;
    float s = 0.f, ss = 0.f;
    for (int i = threadIdx.x; i < HW; i += 256) { float v = x[i]; s += v; ss += v * v; }
    sh1[threadIdx.x] = s; sh2[threadIdx.x] = ss;
    __syncthreads();
    for (int off = 128; off > 0; off >>= 1) {
        if (threadIdx.x < off) {
            sh1[threadIdx.x] += sh1[threadIdx.x + off];
            sh2[threadIdx.x] += sh2[threadIdx.x + off];
        }
        __syncthreads();
    }
    float m   = sh1[0] / (float)HW;
    float var = sh2[0] / (float)HW - m * m;
    float rsd = rsqrtf(fmaxf(var, 0.f) + EPS);
    for (int i = threadIdx.x; i < HW; i += 256) {
        float v = (x[i] - m) * rsd;
        y[i] = v >= 0.f ? v : 0.2f * v;
    }
}

// mean over spatial per (n,c)
__global__ void spatial_mean_kernel(const float* __restrict__ in,
                                    float* __restrict__ mean, int HW) {
    __shared__ float sh[256];
    const float* x = in + (size_t)blockIdx.x * HW;
    float s = 0.f;
    for (int i = threadIdx.x; i < HW; i += 256) s += x[i];
    sh[threadIdx.x] = s;
    __syncthreads();
    for (int off = 128; off > 0; off >>= 1) {
        if (threadIdx.x < off) sh[threadIdx.x] += sh[threadIdx.x + off];
        __syncthreads();
    }
    if (threadIdx.x == 0) mean[blockIdx.x] = sh[0] / (float)HW;
}

// ---------------------------------------------------------------------------
// Correlation prep: subtract spatial mean, L2-normalize over 64 channels per
// position, emit f16 in position-major layout [n][pos][64].
__global__ void prep16_kernel(const float* __restrict__ x,      // [2,64,4096]
                              const float* __restrict__ meanNC, // [128]
                              _Float16* __restrict__ o) {       // [2,4096,64]
    int tid = blockIdx.x * 256 + threadIdx.x;  // n*4096 + p
    int n = tid >> 12;
    int p = tid & 4095;
    float v[64];
    float ss = 0.f;
    #pragma unroll
    for (int c = 0; c < 64; ++c) {
        float t = x[((size_t)(n * 64 + c) << 12) + p] - meanNC[n * 64 + c];
        v[c] = t; ss += t * t;
    }
    float sc = 1.0f / (sqrtf(ss) + EPS);
    _Float16* d = o + (size_t)tid * 64;
    #pragma unroll
    for (int c = 0; c < 64; ++c) d[c] = (_Float16)(v[c] * sc);
}

// ---------------------------------------------------------------------------
// Fused correlation: energy (WMMA) -> column softmax (axis p) -> fc warp.
// One block owns 16 columns q over the FULL p range; 8 waves split p.
// Two WMMA passes: pass 1 gathers per-column (max, sum-exp) deterministically
// via LDS; pass 2 recomputes the identical tiles, streams the normalized
// weights to corr with non-temporal stores (written once, never re-read) and
// accumulates fcw[n,c,q] = sum_p fc[n,c,p]*softmax in registers.
union AFrag { v16h v; v8h h[2]; };

__device__ __forceinline__ v8f energy_tile(const _Float16* __restrict__ fbp,
                                           int half, v16h b0, v16h b1) {
    AFrag A0, A1;
    A0.h[0] = *(const v8h*)(fbp + half * 16);
    A0.h[1] = *(const v8h*)(fbp + 8 + half * 16);
    A1.h[0] = *(const v8h*)(fbp + 32 + half * 16);
    A1.h[1] = *(const v8h*)(fbp + 40 + half * 16);
    v8f c = {};
    c = __builtin_amdgcn_wmma_f32_16x16x32_f16(false, A0.v, false, b0,
                                               (short)0, c, false, false);
    c = __builtin_amdgcn_wmma_f32_16x16x32_f16(false, A1.v, false, b1,
                                               (short)0, c, false, false);
    return c;
}

__global__ void fused_corr_kernel(const _Float16* __restrict__ fa16,
                                  const _Float16* __restrict__ fb16,
                                  const float* __restrict__ fc_small, // [2,3,4096]
                                  float* __restrict__ corr,           // [2,4096,4096]
                                  float* __restrict__ fcw) {          // [2,3,4096]
    __shared__ float smx[256], ssm[256];
    __shared__ float sacc[3][256];
    int wave = threadIdx.x >> 5;          // 0..7 : p-range slice
    int lane = threadIdx.x & 31;
    int half = lane >> 4;                 // row-half of the 16x16 D tile
    int m    = lane & 15;                 // column within tile / A row
    int n    = blockIdx.y;
    int qBase = blockIdx.x * 16;
    int q     = qBase + m;

    // B fragment (columns) is invariant over the whole p loop
    const _Float16* fap = fa16 + ((size_t)(n * 4096 + q)) * 64;
    v16h b0 = *(const v16h*)(fap + half * 16);
    v16h b1 = *(const v16h*)(fap + 32 + half * 16);
    const _Float16* fbBase = fb16 + (size_t)n * 4096 * 64;

    int pt0 = wave * 32, pt1 = pt0 + 32;  // 32 tiles of 16 rows = 512 p per wave

    // ---- pass 1: per-lane online (max, sum-exp) over this lane's rows
    float mx = -3.0e38f, s = 0.f;
    for (int pt = pt0; pt < pt1; ++pt) {
        v8f c = energy_tile(fbBase + (size_t)(pt * 16 + m) * 64, half, b0, b1);
        float e[8], tmax = -3.0e38f;
        #pragma unroll
        for (int r = 0; r < 8; ++r) { e[r] = ALPHA * c[r]; tmax = fmaxf(tmax, e[r]); }
        if (tmax > mx) { s *= __expf(mx - tmax); mx = tmax; }
        #pragma unroll
        for (int r = 0; r < 8; ++r) s += __expf(e[r] - mx);
    }
    smx[threadIdx.x] = mx; ssm[threadIdx.x] = s;
    __syncthreads();
    // combine 8 waves x 2 halves for this lane's column m (deterministic order)
    float gm = -3.0e38f;
    #pragma unroll
    for (int i = 0; i < 16; ++i)
        gm = fmaxf(gm, smx[(i >> 1) * 32 + (i & 1) * 16 + m]);
    float gs = 0.f;
    #pragma unroll
    for (int i = 0; i < 16; ++i) {
        int idx = (i >> 1) * 32 + (i & 1) * 16 + m;
        gs += ssm[idx] * __expf(smx[idx] - gm);
    }
    float rs = 1.0f / gs;

    // ---- pass 2: recompute (bit-identical), normalize, stream out, warp-acc
    float a0 = 0.f, a1 = 0.f, a2 = 0.f;
    const float* fcb = fc_small + (size_t)n * 3 * 4096;
    for (int pt = pt0; pt < pt1; ++pt) {
        v8f c = energy_tile(fbBase + (size_t)(pt * 16 + m) * 64, half, b0, b1);
        float w[8];
        #pragma unroll
        for (int r = 0; r < 8; ++r) w[r] = __expf(ALPHA * c[r] - gm) * rs;
        int po = pt * 16 + half * 8;                       // first p row of this lane
        size_t rowBase = ((size_t)(n * 4096 + po)) * 4096 + q;
        #pragma unroll
        for (int r = 0; r < 8; ++r)
            __builtin_nontemporal_store(w[r], corr + rowBase + ((size_t)r << 12));
        const float4* f0 = (const float4*)(fcb + po);
        const float4* f1 = (const float4*)(fcb + 4096 + po);
        const float4* f2 = (const float4*)(fcb + 8192 + po);
        float4 u, v;
        u = f0[0]; v = f0[1];
        a0 += u.x*w[0] + u.y*w[1] + u.z*w[2] + u.w*w[3]
            + v.x*w[4] + v.y*w[5] + v.z*w[6] + v.w*w[7];
        u = f1[0]; v = f1[1];
        a1 += u.x*w[0] + u.y*w[1] + u.z*w[2] + u.w*w[3]
            + v.x*w[4] + v.y*w[5] + v.z*w[6] + v.w*w[7];
        u = f2[0]; v = f2[1];
        a2 += u.x*w[0] + u.y*w[1] + u.z*w[2] + u.w*w[3]
            + v.x*w[4] + v.y*w[5] + v.z*w[6] + v.w*w[7];
    }
    sacc[0][threadIdx.x] = a0; sacc[1][threadIdx.x] = a1; sacc[2][threadIdx.x] = a2;
    __syncthreads();
    if (threadIdx.x < 16) {
        #pragma unroll
        for (int ch = 0; ch < 3; ++ch) {
            float t = 0.f;
            #pragma unroll
            for (int i = 0; i < 16; ++i)
                t += sacc[ch][(i >> 1) * 32 + (i & 1) * 16 + threadIdx.x];
            fcw[((size_t)(n * 3 + ch) << 12) + qBase + threadIdx.x] = t;
        }
    }
}

// ---------------------------------------------------------------------------
// Reflection-pad(1) + 3x3 conv stride 2, 3->3 channels.
__global__ void conv3x3_reflect_s2_kernel(const float* __restrict__ in,
                                          const float* __restrict__ W9,
                                          const float* __restrict__ bias,
                                          float* __restrict__ out,
                                          int Hin, int Win) {
    int Ho = Hin >> 1, Wo = Win >> 1;
    int tid = blockIdx.x * 256 + threadIdx.x;
    int total = 2 * 3 * Ho * Wo;
    if (tid >= total) return;
    int x  = tid % Wo;
    int y  = (tid / Wo) % Ho;
    int co = (tid / (Wo * Ho)) % 3;
    int n  = tid / (3 * Wo * Ho);
    float acc = bias[co];
    for (int ci = 0; ci < 3; ++ci) {
        const float* src = in + (size_t)(n * 3 + ci) * Hin * Win;
        #pragma unroll
        for (int dy = 0; dy < 3; ++dy) {
            int yy = 2 * y + dy - 1;
            yy = yy < 0 ? -yy : (yy >= Hin ? 2 * Hin - 2 - yy : yy);
            #pragma unroll
            for (int dx = 0; dx < 3; ++dx) {
                int xx = 2 * x + dx - 1;
                xx = xx < 0 ? -xx : (xx >= Win ? 2 * Win - 2 - xx : xx);
                acc += W9[((co * 3 + ci) * 3 + dy) * 3 + dx] * src[(size_t)yy * Win + xx];
            }
        }
    }
    out[tid] = acc;
}

// ---------------------------------------------------------------------------
// Bilinear 2x upsample (align_corners) + zero-pad(1) + 3x3 conv, 3->3 ch.
__global__ void up2_conv3x3_kernel(const float* __restrict__ in,
                                   const float* __restrict__ W9,
                                   const float* __restrict__ bias,
                                   float* __restrict__ out,
                                   int H, int Wd) {
    int Ho = 2 * H, Wo = 2 * Wd;
    int tid = blockIdx.x * 256 + threadIdx.x;
    int total = 2 * 3 * Ho * Wo;
    if (tid >= total) return;
    int x  = tid % Wo;
    int y  = (tid / Wo) % Ho;
    int co = (tid / (Wo * Ho)) % 3;
    int n  = tid / (3 * Wo * Ho);
    float acc = bias[co];
    #pragma unroll
    for (int dy = 0; dy < 3; ++dy) {
        int yy = y + dy - 1;
        if (yy < 0 || yy >= Ho) continue;               // zero pad
        float py = (float)(yy * (H - 1)) / (float)(Ho - 1);
        int iy0 = (int)floorf(py);
        float fy = py - (float)iy0;
        int iy1 = iy0 + 1 < H ? iy0 + 1 : H - 1;
        #pragma unroll
        for (int dx = 0; dx < 3; ++dx) {
            int xx = x + dx - 1;
            if (xx < 0 || xx >= Wo) continue;
            float px = (float)(xx * (Wd - 1)) / (float)(Wo - 1);
            int ix0 = (int)floorf(px);
            float fx = px - (float)ix0;
            int ix1 = ix0 + 1 < Wd ? ix0 + 1 : Wd - 1;
            for (int ci = 0; ci < 3; ++ci) {
                const float* s = in + (size_t)(n * 3 + ci) * H * Wd;
                float v00 = s[(size_t)iy0 * Wd + ix0], v01 = s[(size_t)iy0 * Wd + ix1];
                float v10 = s[(size_t)iy1 * Wd + ix0], v11 = s[(size_t)iy1 * Wd + ix1];
                float vt = v00 * (1.f - fy) + v10 * fy;   // height first (matches ref)
                float vb = v01 * (1.f - fy) + v11 * fy;
                float v  = vt * (1.f - fx) + vb * fx;
                acc += W9[((co * 3 + ci) * 3 + dy) * 3 + dx] * v;
            }
        }
    }
    out[tid] = acc;
}

// ---------------------------------------------------------------------------
extern "C" void kernel_launch(void* const* d_in, const int* in_sizes, int n_in,
                              void* d_out, int out_size, void* d_ws, size_t ws_size,
                              hipStream_t stream) {
    const float* fa_raw = (const float*)d_in[0];
    const float* fb_raw = (const float*)d_in[1];
    const float* fc_raw = (const float*)d_in[2];
    const float* Wa  = (const float*)d_in[3];
    const float* ba  = (const float*)d_in[4];
    const float* Wb  = (const float*)d_in[5];
    const float* bb  = (const float*)d_in[6];
    const float* Wc1 = (const float*)d_in[7];
    const float* bc1 = (const float*)d_in[8];
    const float* Wc2 = (const float*)d_in[9];
    const float* bc2 = (const float*)d_in[10];
    const float* Wu1 = (const float*)d_in[11];
    const float* bu1 = (const float*)d_in[12];
    const float* Wu2 = (const float*)d_in[13];
    const float* bu2 = (const float*)d_in[14];

    float* out_warp = (float*)d_out;                 // [2,3,256,256]
    float* corr     = out_warp + 2 * 3 * 256 * 256;  // [2,4096,4096]

    char* w = (char*)d_ws;
    size_t o = 0;
    auto carve = [&](size_t bytes) { char* p = w + o; o += (bytes + 255) & ~(size_t)255; return p; };
    float*    faBuf    = (float*)   carve(2u * 64 * 4096 * 4);
    float*    fbBuf    = (float*)   carve(2u * 64 * 4096 * 4);
    float*    meanA    = (float*)   carve(128 * 4);
    float*    meanB    = (float*)   carve(128 * 4);
    _Float16* fa16     = (_Float16*)carve(2u * 4096 * 64 * 2);
    _Float16* fb16     = (_Float16*)carve(2u * 4096 * 64 * 2);
    float*    c1buf    = (float*)   carve(2u * 3 * 128 * 128 * 4);
    float*    fc_small = (float*)   carve(2u * 3 * 4096 * 4);
    float*    fcw      = (float*)   carve(2u * 3 * 4096 * 4);
    float*    u1buf    = (float*)   carve(2u * 3 * 128 * 128 * 4);
    float*    u2buf    = (float*)   carve(2u * 3 * 256 * 256 * 4);

    // --- fa branch: 1x1 conv -> IN+leaky (in-place) -> spatial mean -> f16 prep
    conv1x1_kernel<<<2048, 256, 0, stream>>>(fa_raw, Wa, ba, faBuf);
    inorm_leaky_kernel<<<128, 256, 0, stream>>>(faBuf, faBuf, 4096);
    spatial_mean_kernel<<<128, 256, 0, stream>>>(faBuf, meanA, 4096);
    prep16_kernel<<<32, 256, 0, stream>>>(faBuf, meanA, fa16);

    // --- fb branch
    conv1x1_kernel<<<2048, 256, 0, stream>>>(fb_raw, Wb, bb, fbBuf);
    inorm_leaky_kernel<<<128, 256, 0, stream>>>(fbBuf, fbBuf, 4096);
    spatial_mean_kernel<<<128, 256, 0, stream>>>(fbBuf, meanB, 4096);
    prep16_kernel<<<32, 256, 0, stream>>>(fbBuf, meanB, fb16);

    // --- fc branch: two reflect-pad stride-2 convs + IN+leaky
    conv3x3_reflect_s2_kernel<<<(2 * 3 * 128 * 128) / 256, 256, 0, stream>>>(
        fc_raw, Wc1, bc1, c1buf, 256, 256);
    inorm_leaky_kernel<<<6, 256, 0, stream>>>(c1buf, c1buf, 128 * 128);
    conv3x3_reflect_s2_kernel<<<(2 * 3 * 64 * 64) / 256, 256, 0, stream>>>(
        c1buf, Wc2, bc2, fc_small, 128, 128);
    inorm_leaky_kernel<<<6, 256, 0, stream>>>(fc_small, fc_small, 64 * 64);

    // --- fused WMMA energy + column softmax + warp (corr written exactly once)
    fused_corr_kernel<<<dim3(256, 2), 256, 0, stream>>>(fa16, fb16, fc_small, corr, fcw);

    // --- upsample blocks: up2x + zero-pad conv + IN+leaky, twice
    up2_conv3x3_kernel<<<(2 * 3 * 128 * 128) / 256, 256, 0, stream>>>(
        fcw, Wu1, bu1, u1buf, 64, 64);
    inorm_leaky_kernel<<<6, 256, 0, stream>>>(u1buf, u1buf, 128 * 128);
    up2_conv3x3_kernel<<<(2 * 3 * 256 * 256) / 256, 256, 0, stream>>>(
        u1buf, Wu2, bu2, u2buf, 128, 128);
    inorm_leaky_kernel<<<6, 256, 0, stream>>>(u2buf, out_warp, 256 * 256);
}
